// LetheAttention_17257178596119
// MI455X (gfx1250) — matrix operand order repaired and checked by reading, compiled, not run on hardware
//
#include <hip/hip_runtime.h>
#include <hip/hip_bf16.h>

// ---------------- problem constants ----------------
constexpr int cNH = 16;    // heads
constexpr int cDH = 128;   // head dim
constexpr int cROT = 32;   // rope dims
constexpr int cNN = 32;    // knn neighbors (== wave32!)
constexpr int cM = 8192;   // memory slots
constexpr int cS = 1024;   // sequence
constexpr int cH = 2048;   // hidden

typedef __bf16 bf16_t;
typedef __attribute__((ext_vector_type(16))) __bf16 v16bf;
typedef __attribute__((ext_vector_type(8)))  __bf16 v8bf;
typedef __attribute__((ext_vector_type(8)))  float  v8f;
typedef __attribute__((ext_vector_type(4)))  unsigned int v4u;
typedef __attribute__((ext_vector_type(8)))  int v8i;
typedef __attribute__((ext_vector_type(4)))  int v4i;

// Tensor Data Mover availability (this toolchain: 6-arg builtin,
// (uint32x4 g0, int32x8 g1, int32x4 g2, int32x4 g3, int32x8 g4, i32 cpol))
#if defined(__has_builtin)
#if __has_builtin(__builtin_amdgcn_tensor_load_to_lds) && \
    __has_builtin(__builtin_amdgcn_s_wait_tensorcnt)
#define USE_TDM 1
#endif
#endif
#ifndef USE_TDM
#define USE_TDM 0
#endif

// ---------------- WMMA helpers ----------------
__device__ inline v8f wmma_bf16(v16bf a, v16bf b, v8f c) {
  // D = A(16x32 bf16) * B(32x16 bf16) + C(16x16 f32)
  return __builtin_amdgcn_wmma_f32_16x16x32_bf16(
      /*neg_a=*/false, a, /*neg_b=*/false, b,
      /*c_mod=*/(short)0, c, /*reuse_a=*/false, /*reuse_b=*/false);
}

__device__ inline v16bf cvt16(const float* __restrict__ x) {
  v16bf f;
#pragma unroll
  for (int e = 0; e < 16; ++e) f[e] = (__bf16)x[e];
  return f;
}
__device__ inline v16bf cat8(v8bf a, v8bf b) {
  return __builtin_shufflevector(a, b, 0, 1, 2, 3, 4, 5, 6, 7, 8, 9, 10, 11, 12,
                                 13, 14, 15);
}

// A-operand fragment (row = lane%16). 16-bit A 16x32 layout:
// lanes 0-15: K={k0+0..7, k0+16..23}; lanes 16-31: K={k0+8..15, k0+24..31}
__device__ inline v16bf a_frag_f32(const float* rowp, int k0, int hi) {
  float t[16];
  const float* b = rowp + k0 + (hi ? 8 : 0);
  *(float4*)(t + 0)  = *(const float4*)(b + 0);
  *(float4*)(t + 4)  = *(const float4*)(b + 4);
  *(float4*)(t + 8)  = *(const float4*)(b + 16);
  *(float4*)(t + 12) = *(const float4*)(b + 20);
  return cvt16(t);
}
__device__ inline v16bf a_frag_bf16(const bf16_t* __restrict__ rowp, int k0, int hi) {
  const bf16_t* p = rowp + k0 + (hi ? 8 : 0);
  v8bf x0 = *(const v8bf*)(p);
  v8bf x1 = *(const v8bf*)(p + 16);
  return cat8(x0, x1);
}

// B-operand fragment for NT source (rowp = row n of B[N,K]; col = lane%16).
// 16-bit B 32x16 layout: lanes 0-15: K={k0+0..7, k0+16..23}; lanes 16-31: K={k0+8..15, k0+24..31}
__device__ inline v16bf b_frag_f32(const float* __restrict__ rowp, int k0, int hi) {
  float t[16];
  const float4* p = (const float4*)(rowp + k0 + hi * 8);
  *(float4*)(t + 0)  = p[0];
  *(float4*)(t + 4)  = p[1];
  *(float4*)(t + 8)  = p[4];
  *(float4*)(t + 12) = p[5];
  return cvt16(t);
}
__device__ inline v16bf b_frag_bf16(const bf16_t* __restrict__ rowp, int k0, int hi) {
  const bf16_t* p = rowp + k0 + hi * 8;
  v8bf x0 = *(const v8bf*)(p);
  v8bf x1 = *(const v8bf*)(p + 16);
  return cat8(x0, x1);
}

#if USE_TDM
// 2D TDM tile load: tile_rows x tile_k f32 elements, row stride = strideK elems,
// packed contiguously into LDS at lds_off. Descriptor per CDNA5 ISA section 8.
__device__ inline void tdm_load_tile_2d(unsigned lds_off, const float* gsrc,
                                        int tile_k, int tile_rows,
                                        unsigned rem_k, unsigned rem_rows,
                                        unsigned strideK) {
  unsigned long long ga = (unsigned long long)(__UINTPTR_TYPE__)gsrc;
  v4u g0;
  g0[0] = 1u;                                    // count=1, user D#
  g0[1] = lds_off;                               // lds_addr (bytes)
  g0[2] = (unsigned)(ga & 0xffffffffu);          // global_addr[31:0]
  g0[3] = (unsigned)((ga >> 32) & 0x01ffffffu) | (2u << 30);  // addr[56:32]|type=2
  v8i g1;
  g1[0] = (int)(2u << 16);                       // data_size=4B; no mask/pad/iter
  g1[1] = (int)((rem_k & 0xffffu) << 16);        // tensor_dim0 lo16 @ [63:48]
  g1[2] = (int)(((rem_k >> 16) & 0xffffu) | ((rem_rows & 0xffffu) << 16));
  g1[3] = (int)(((rem_rows >> 16) & 0xffffu) | ((unsigned)tile_k << 16));
  g1[4] = (int)((unsigned)tile_rows & 0xffffu);  // tile_dim1; tile_dim2=0
  g1[5] = (int)strideK;                          // tensor_dim0_stride lo32
  g1[6] = 0;                                     // stride hi16 | dim1_stride lo16
  g1[7] = 0;
  v4i z4 = {0, 0, 0, 0};
  v8i z8 = {0, 0, 0, 0, 0, 0, 0, 0};
  __builtin_amdgcn_tensor_load_to_lds(g0, g1, z4, z4, z8, 0);
}
#endif

// ---------------- Kernel 1/5: NT GEMM, C = A*B^T + bias ----------------
// A[Mr,K], B[N,K], C[Mr,N] f32. Block = 4 waves, 64x64 output tile.
// A-tile (64x32) staged in LDS (TDM when available); wave computes 4 16x16
// tiles sharing one B fragment per k-step (4 WMMA / k-step / wave).
__global__ __launch_bounds__(128) void gemm_nt_bias(
    const float* __restrict__ A, const float* __restrict__ Bm,
    const float* __restrict__ bias, float* __restrict__ C,
    int Mr, int N, int K) {
  __shared__ __align__(16) float Atile[64 * 32];  // 8 KB
  const int wave = threadIdx.x >> 5;
  const int lane = threadIdx.x & 31;
  const int hi = lane >> 4;
  const int lr = lane & 15;
  const int m0 = blockIdx.y * 64;
  const int n0 = blockIdx.x * 64 + wave * 16;
  const float* brow = Bm + (size_t)(n0 + lr) * K;
  v8f acc[4] = {{}, {}, {}, {}};
  for (int k0 = 0; k0 < K; k0 += 32) {
    __syncthreads();  // previous-iteration readers done before overwrite
#if USE_TDM
    if (wave == 0) {
      tdm_load_tile_2d((unsigned)(__UINTPTR_TYPE__)&Atile[0],
                       A + (size_t)m0 * K + k0, /*tile_k=*/32, /*tile_rows=*/64,
                       (unsigned)(K - k0), (unsigned)(Mr - m0), (unsigned)K);
      __builtin_amdgcn_s_wait_tensorcnt(0);
    }
#else
    for (int i = threadIdx.x; i < 64 * 32; i += 128) {
      int r = i >> 5, c = i & 31;
      Atile[i] = A[(size_t)(m0 + r) * K + k0 + c];
    }
#endif
    __syncthreads();
    v16bf bf = b_frag_f32(brow, k0, hi);
#pragma unroll
    for (int mt = 0; mt < 4; ++mt) {
      float t[16];
      const float* ar = Atile + (mt * 16 + lr) * 32 + (hi ? 8 : 0);
      *(float4*)(t + 0)  = *(const float4*)(ar + 0);
      *(float4*)(t + 4)  = *(const float4*)(ar + 4);
      *(float4*)(t + 8)  = *(const float4*)(ar + 16);
      *(float4*)(t + 12) = *(const float4*)(ar + 20);
      acc[mt] = wmma_bf16(cvt16(t), bf, acc[mt]);
    }
  }
  const int col = n0 + lr;
  const float bv = bias[col];
#pragma unroll
  for (int mt = 0; mt < 4; ++mt)
#pragma unroll
    for (int j = 0; j < 8; ++j) {
      int m = m0 + mt * 16 + j + hi * 8;  // D: row = j + 8*hi, col = lane%16
      C[(size_t)m * N + col] = acc[mt][j] + bv;
    }
}

// ---------------- Kernel 2: l2norm + RoPE + layout ----------------
// qkv[s][h][3*DH] -> qf/qb/kb [h][s][d], vtb [h][d][s] (bf16, transposed V)
__global__ __launch_bounds__(32) void norm_rope_kernel(
    const float* __restrict__ qkv, float* __restrict__ qf,
    bf16_t* __restrict__ qb, bf16_t* __restrict__ kb, bf16_t* __restrict__ vtb) {
  const int s = blockIdx.x, h = blockIdx.y, lane = threadIdx.x;
  const float* base = qkv + ((size_t)s * cNH + h) * (3 * cDH);
  float cs[4], sn[4];
#pragma unroll
  for (int e = 0; e < 4; ++e) {
    int d = lane * 4 + e;
    int i = d & 15;                                   // freq index (emb = [ang,ang])
    float inv = __powf(10000.0f, -(float)i / 16.0f);  // BASE^(-2i/ROT)
    float ang = (float)s * inv;                       // position_ids == arange
    cs[e] = __cosf(ang);
    sn[e] = __sinf(ang);
  }
#pragma unroll
  for (int which = 0; which < 2; ++which) {  // 0 = q, 1 = k
    const float* src = base + which * cDH;
    float x[4];
    float ss = 0.f;
#pragma unroll
    for (int e = 0; e < 4; ++e) { x[e] = src[lane * 4 + e]; ss += x[e] * x[e]; }
#pragma unroll
    for (int off = 16; off; off >>= 1) ss += __shfl_xor(ss, off);
    float scl = 1.0f / fmaxf(sqrtf(ss), 1e-12f);
#pragma unroll
    for (int e = 0; e < 4; ++e) x[e] *= scl;
    // RoPE on dims < 32: partner dim d^16 lives 4 lanes away, same element slot
#pragma unroll
    for (int e = 0; e < 4; ++e) {
      float p = __shfl_xor(x[e], 4);
      int d = lane * 4 + e;
      if (d < cROT) {
        float rh = (d < cROT / 2) ? -p : p;
        x[e] = x[e] * cs[e] + rh * sn[e];
      }
    }
    size_t o = ((size_t)h * cS + s) * cDH + lane * 4;
    if (which == 0) {
#pragma unroll
      for (int e = 0; e < 4; ++e) { qf[o + e] = x[e]; qb[o + e] = (bf16_t)x[e]; }
    } else {
#pragma unroll
      for (int e = 0; e < 4; ++e) kb[o + e] = (bf16_t)x[e];
    }
  }
  const float* vsrc = base + 2 * cDH;
#pragma unroll
  for (int e = 0; e < 4; ++e) {
    int d = lane * 4 + e;
    vtb[((size_t)h * cDH + d) * cS + s] = (bf16_t)vsrc[d];
  }
}

// ---------------- Kernel 3: memory scores + top-32 ----------------
// One wave per (h,s). Cooperative dot per key; NN==32 so each lane owns exactly
// one slot of the top-list; insertion = wave argmin + conditional replace.
__global__ __launch_bounds__(32) void topk_kernel(
    const float* __restrict__ qf, const float* __restrict__ mem_keys,
    int* __restrict__ topidx) {
  const int s = blockIdx.x, h = blockIdx.y, lane = threadIdx.x;
  const float* q = qf + ((size_t)h * cS + s) * cDH + lane * 4;
  const float4 q4 = *(const float4*)q;
  float bestv = -3.0e38f;
  int besti = 0;
  const float* kbase = mem_keys + (size_t)h * cM * cDH;
  for (int m = 0; m < cM; ++m) {
    const float* kr = kbase + (size_t)m * cDH + lane * 4;
    __builtin_prefetch(kr + 16 * cDH, 0, 1);  // global_prefetch_b8 (L2-resident)
    const float4 k4 = *(const float4*)kr;
    float part = q4.x * k4.x + q4.y * k4.y + q4.z * k4.z + q4.w * k4.w;
#pragma unroll
    for (int off = 16; off; off >>= 1) part += __shfl_xor(part, off);
    // wave-wide (min, argmin-lane) of current top-list
    float mv = bestv;
    int ml = lane;
#pragma unroll
    for (int off = 16; off; off >>= 1) {
      float ov = __shfl_xor(mv, off);
      int ol = __shfl_xor(ml, off);
      if (ov < mv || (ov == mv && ol < ml)) { mv = ov; ml = ol; }
    }
    if (part > mv && lane == ml) { bestv = part; besti = m; }
  }
  topidx[((size_t)h * cS + s) * cNN + lane] = besti;
}

// ---------------- Kernel 4: fused attention per (head, 16-query block) --------
__global__ __launch_bounds__(128) void attn_kernel(
    const float* __restrict__ qf, const bf16_t* __restrict__ qb,
    const bf16_t* __restrict__ kb, const bf16_t* __restrict__ vtb,
    const float* __restrict__ mem_keys, const float* __restrict__ mem_values,
    const int* __restrict__ topidx, const float* __restrict__ amask,
    const float* __restrict__ scale_param, float* __restrict__ attn_out) {
  extern __shared__ float W[];  // [16][LW]  (cols 0..31 = mem, 32.. = local)
  const int LW = cNN + cS;
  const int h = blockIdx.y;
  const int qbase = blockIdx.x * 16;
  const int tid = threadIdx.x, wave = tid >> 5, lane = tid & 31;
  const int hi = lane >> 4, lr = lane & 15;
  const float scale = __expf(scale_param[h]);

  // q A-fragments live in registers for the whole kernel (4 k-steps of 32)
  const bf16_t* qrow = qb + ((size_t)h * cS + qbase + lr) * cDH;
  v16bf aq[4];
#pragma unroll
  for (int t = 0; t < 4; ++t) aq[t] = a_frag_bf16(qrow, t * 32, hi);

  // ---- local scores QK^T (WMMA), causal mask + scale + attention_mask ----
  for (int kt = wave; kt < cS / 16; kt += 4) {
    const bf16_t* krow = kb + ((size_t)h * cS + kt * 16 + lr) * cDH;
    v8f acc = {};
#pragma unroll
    for (int t = 0; t < 4; ++t)
      acc = wmma_bf16(aq[t], b_frag_bf16(krow, t * 32, hi), acc);
    const int kg = kt * 16 + lr;
    const float am = amask[kg];
#pragma unroll
    for (int j = 0; j < 8; ++j) {
      int row = j + hi * 8;
      int qg = qbase + row;
      W[row * LW + cNN + kg] = (kg <= qg) ? acc[j] * scale + am : -3.0e38f;
    }
  }

  // ---- mem scores: q . gathered knn keys (8 threads per query, 4 each) ----
  {
    const int qi = tid >> 3;
    const float* qv = qf + ((size_t)h * cS + qbase + qi) * cDH;
    const int* ip = topidx + ((size_t)h * cS + qbase + qi) * cNN;
#pragma unroll
    for (int nn = 0; nn < 4; ++nn) {
      int n = (tid & 7) + nn * 8;
      const float* kv = mem_keys + ((size_t)h * cM + ip[n]) * cDH;
      float dotv = 0.f;
      for (int d = 0; d < cDH; d += 4) {
        const float4 a4 = *(const float4*)(qv + d);
        const float4 b4 = *(const float4*)(kv + d);
        dotv += a4.x * b4.x + a4.y * b4.y + a4.z * b4.z + a4.w * b4.w;
      }
      W[qi * LW + n] = dotv * scale;
    }
  }
  __syncthreads();

  // ---- softmax over 32 + 1024 per query row ----
  if (tid < 16) {
    float* row = W + tid * LW;
    float mx = -3.0e38f;
    for (int i = 0; i < LW; ++i) mx = fmaxf(mx, row[i]);
    float sum = 0.f;
    for (int i = 0; i < LW; ++i) { float e = __expf(row[i] - mx); row[i] = e; sum += e; }
    float inv = 1.0f / sum;
    for (int i = 0; i < LW; ++i) row[i] *= inv;
  }
  __syncthreads();

  // ---- output: wl @ V (WMMA, bf16 A-frags straight from LDS) + wm @ knn_v ----
#pragma unroll
  for (int c = 0; c < 2; ++c) {
    const int ct = wave + c * 4;  // 8 dim-tiles of 16 across 4 waves x 2
    const int dim = ct * 16 + lr;
    const bf16_t* vrow = vtb + ((size_t)h * cDH + dim) * cS;  // contiguous in K=s
    v8f oacc = {};
    for (int ks = 0; ks < cS; ks += 32) {
      float t[16];
      const float* wrow = W + lr * LW + cNN + ks + (hi ? 8 : 0);
      *(float4*)(t + 0)  = *(const float4*)(wrow + 0);
      *(float4*)(t + 4)  = *(const float4*)(wrow + 4);
      *(float4*)(t + 8)  = *(const float4*)(wrow + 16);
      *(float4*)(t + 12) = *(const float4*)(wrow + 20);
      oacc = wmma_bf16(cvt16(t), b_frag_bf16(vrow, ks, hi), oacc);
    }
#pragma unroll
    for (int j = 0; j < 8; ++j) {
      int row = j + hi * 8;
      int qg = qbase + row;
      const int* ip = topidx + ((size_t)h * cS + qg) * cNN;
      const float* wr = W + row * LW;
      float memsum = 0.f;
      for (int n = 0; n < cNN; ++n)
        memsum += wr[n] * mem_values[((size_t)h * cM + ip[n]) * cDH + dim];
      attn_out[((size_t)qg * cNH + h) * cDH + dim] = oacc[j] + memsum;  // [s][h][d]
    }
  }
}

// ---------------- launch ----------------
extern "C" void kernel_launch(void* const* d_in, const int* in_sizes, int n_in,
                              void* d_out, int out_size, void* d_ws, size_t ws_size,
                              hipStream_t stream) {
  (void)in_sizes; (void)n_in; (void)out_size; (void)ws_size;
  const float* hidden      = (const float*)d_in[0];
  const float* amask       = (const float*)d_in[1];
  /* d_in[2] position_ids == arange(S): folded into RoPE kernel */
  const float* Wqkv        = (const float*)d_in[3];
  const float* bqkv        = (const float*)d_in[4];
  const float* Wd          = (const float*)d_in[5];
  const float* bd          = (const float*)d_in[6];
  const float* scale_param = (const float*)d_in[7];
  const float* mem_keys    = (const float*)d_in[8];
  const float* mem_values  = (const float*)d_in[9];
  float* out = (float*)d_out;

  char* ws = (char*)d_ws;
  auto alloc = [&](size_t bytes) {
    char* p = ws;
    ws += (bytes + 255) & ~(size_t)255;
    return (void*)p;
  };
  float*  qkv  = (float*)alloc((size_t)cS * 3 * cH * sizeof(float));       // 24 MB
  float*  qf   = (float*)alloc((size_t)cNH * cS * cDH * sizeof(float));    //  8 MB
  bf16_t* qb   = (bf16_t*)alloc((size_t)cNH * cS * cDH * sizeof(bf16_t));  //  4 MB
  bf16_t* kb   = (bf16_t*)alloc((size_t)cNH * cS * cDH * sizeof(bf16_t));  //  4 MB
  bf16_t* vtb  = (bf16_t*)alloc((size_t)cNH * cDH * cS * sizeof(bf16_t));  //  4 MB
  int*    topi = (int*)alloc((size_t)cNH * cS * cNN * sizeof(int));        //  2 MB
  float*  attn = (float*)alloc((size_t)cS * cH * sizeof(float));           //  8 MB

  // 1) qkv = hidden @ Wqkv^T + bqkv     (M=1024, N=6144, K=2048)
  gemm_nt_bias<<<dim3(3 * cH / 64, cS / 64), 128, 0, stream>>>(
      hidden, Wqkv, bqkv, qkv, cS, 3 * cH, cH);
  // 2) l2norm + rope + bf16/layout
  norm_rope_kernel<<<dim3(cS, cNH), 32, 0, stream>>>(qkv, qf, qb, kb, vtb);
  // 3) top-32 over 8192 memory keys per (h,s)
  topk_kernel<<<dim3(cS, cNH), 32, 0, stream>>>(qf, mem_keys, topi);
  // 4) fused attention
  attn_kernel<<<dim3(cS / 16, cNH), 128, (size_t)(cNN + cS) * 16 * sizeof(float),
                stream>>>(qf, qb, kb, vtb, mem_keys, mem_values, topi, amask,
                          scale_param, attn);
  // 5) out = attn @ Wd^T + bd          (M=1024, N=2048, K=2048)
  gemm_nt_bias<<<dim3(cH / 64, cS / 64), 128, 0, stream>>>(
      attn, Wd, bd, out, cS, cH, cH);
}
